// LongformerForRegression_62113817035436
// MI455X (gfx1250) — matrix operand order repaired and checked by reading, not compiled, not run on hardware
//
#include <hip/hip_runtime.h>

typedef unsigned short u16;
typedef __attribute__((ext_vector_type(16))) __bf16 v16bf;
typedef __attribute__((ext_vector_type(8)))  float  v8f;
typedef __attribute__((ext_vector_type(4)))  unsigned su4;
typedef __attribute__((ext_vector_type(8)))  unsigned su8;

#define S_LEN 4096
#define HID   768
#define NHEAD 12
#define DHEAD 64
#define FFDIM 3072
#define WWIN  256
#define NEGV  -1000000000.0f

// ---------- bf16 helpers (raw u16 storage, RNE convert) ----------
__device__ __forceinline__ u16 f2bf(float f) {
    unsigned u = __float_as_uint(f);
    unsigned r = u + 0x7FFFu + ((u >> 16) & 1u);
    return (u16)(r >> 16);
}
__device__ __forceinline__ float bf2f(u16 h) {
    return __uint_as_float(((unsigned)h) << 16);
}

union Frag16 { v16bf v; unsigned u[8]; uint4 q[2]; };

__device__ __forceinline__ v8f wmma_bf16(v16bf a, v16bf b, v8f c) {
    return __builtin_amdgcn_wmma_f32_16x16x32_bf16(false, a, false, b, (short)0, c, false, false);
}

// A fragment (16x32 bf16, M=row per lane%16): lane<16 pairs K={0,2,4,6,16,18,20,22}+kbase
// lane>=16 pairs K={8,10,12,14,24,26,28,30}+kbase. Source row-major, stride in elems.
__device__ __forceinline__ v16bf load_afrag16(const u16* base, int stride, int row, int kbase, int lane) {
    Frag16 f;
    const unsigned* rp = (const unsigned*)(base + (size_t)row * stride);
    int k0 = (kbase + ((lane & 16) ? 8 : 0)) >> 1;
#pragma unroll
    for (int p = 0; p < 4; ++p) { f.u[p] = rp[k0 + p]; f.u[p + 4] = rp[k0 + 8 + p]; }
    return f.v;
}

// A fragment built from f32 data (probs) with on-the-fly bf16 convert.
__device__ __forceinline__ v16bf load_afrag_f32(const float* rowp, int kbase, int lane) {
    Frag16 f;
    int b = kbase + ((lane & 16) ? 8 : 0);
#pragma unroll
    for (int p = 0; p < 4; ++p) {
        int k = b + 2 * p;
        f.u[p]     = ((unsigned)f2bf(rowp[k + 1]) << 16) | f2bf(rowp[k]);
        int k2 = b + 16 + 2 * p;
        f.u[p + 4] = ((unsigned)f2bf(rowp[k2 + 1]) << 16) | f2bf(rowp[k2]);
    }
    return f.v;
}

// B fragment (32x16 bf16, N=col per lane%16): lane<16 K={0..7,16..23}+kbase, lane>=16 K={8..15,24..31}+kbase,
// stored K-contiguous per column (stride in elems). 2x 16B LDS loads.
__device__ __forceinline__ v16bf load_bfrag16(const u16* base, int stride, int col, int kbase, int lane) {
    Frag16 f;
    const uint4* p = (const uint4*)(base + (size_t)col * stride + kbase + ((lane & 16) ? 8 : 0));
    f.q[0] = p[0];
    f.q[1] = p[2];
    return f.v;
}

__device__ __forceinline__ unsigned rfl(unsigned x) {
    return (unsigned)__builtin_amdgcn_readfirstlane((int)x);
}

// ---------- f32 -> bf16 convert ----------
__global__ __launch_bounds__(256) void f2bf_kernel(const float* __restrict__ in,
                                                   u16* __restrict__ out, int n) {
    int i = blockIdx.x * 256 + threadIdx.x;
    if (i < n) out[i] = f2bf(in[i]);
}

// ---------- embedding gather: word_emb[id] + pos_emb ----------
__global__ __launch_bounds__(256) void embed_kernel(const int* __restrict__ ids,
                                                    const float* __restrict__ we,
                                                    const float* __restrict__ pe,
                                                    float* __restrict__ out) {
    int s = blockIdx.x;
    int id = ids[s];
#pragma unroll
    for (int i = 0; i < 3; ++i) {
        int d = threadIdx.x + i * 256;
        out[(size_t)s * HID + d] = we[(size_t)id * HID + d] + pe[(size_t)s * HID + d];
    }
}

// ---------- LayerNorm (block per token), writes f32 + bf16 ----------
__global__ __launch_bounds__(256) void ln_kernel(const float* __restrict__ in,
                                                 const float* __restrict__ g,
                                                 const float* __restrict__ b,
                                                 float* __restrict__ xf,
                                                 u16* __restrict__ xb) {
    __shared__ float r1[256], r2[256];
    int s = blockIdx.x, tid = threadIdx.x;
    const float* row = in + (size_t)s * HID;
    float lv[3], sum = 0.f, sq = 0.f;
#pragma unroll
    for (int i = 0; i < 3; ++i) {
        lv[i] = row[tid + i * 256];
        sum += lv[i];
        sq  += lv[i] * lv[i];
    }
    r1[tid] = sum; r2[tid] = sq;
    __syncthreads();
    for (int off = 128; off > 0; off >>= 1) {
        if (tid < off) { r1[tid] += r1[tid + off]; r2[tid] += r2[tid + off]; }
        __syncthreads();
    }
    float mean = r1[0] * (1.f / HID);
    float var  = r2[0] * (1.f / HID) - mean * mean;
    float rstd = rsqrtf(var + 1e-5f);
#pragma unroll
    for (int i = 0; i < 3; ++i) {
        int d = tid + i * 256;
        float y = (lv[i] - mean) * rstd * g[d] + b[d];
        xf[(size_t)s * HID + d] = y;
        xb[(size_t)s * HID + d] = f2bf(y);
    }
}

// ---------- WMMA GEMM: C[M,N] = A[M,K] @ B[K,N] + bias, templated epilogue ----------
// block = 128 threads (4 waves), tile 128x64 (wave: 2 M-strips x 4 N-tiles), K-step 32.
template<bool RESID, bool GELU, bool OUTF32>
__global__ __launch_bounds__(128) void gemm_wmma(const u16* __restrict__ A,
                                                 const u16* __restrict__ B,
                                                 const float* __restrict__ bias,
                                                 const float* __restrict__ resid,
                                                 void* __restrict__ Cout,
                                                 int M, int N, int K) {
    __shared__ u16 sA[128 * 32];  // row-major [128 rows][32 k]
    __shared__ u16 sBt[64 * 32];  // K-transposed [64 n][32 k]
    int tid = threadIdx.x, lane = tid & 31, w = tid >> 5;
    int mb = blockIdx.x * 128, nb = blockIdx.y * 64;
    v8f acc[2][4] = {};

    for (int kb = 0; kb < K; kb += 32) {
        __syncthreads();
#pragma unroll
        for (int i = 0; i < 16; ++i) {          // A tile: 2048 u32
            int idx = tid + i * 128;
            int row = idx >> 4, kp = idx & 15;
            ((unsigned*)sA)[row * 16 + kp] =
                *(const unsigned*)(A + (size_t)(mb + row) * K + kb + kp * 2);
        }
#pragma unroll
        for (int i = 0; i < 8; ++i) {           // B tile, store K-transposed
            int idx = tid + i * 128;
            int kk = idx >> 5, np = idx & 31;
            unsigned uv = *(const unsigned*)(B + (size_t)(kb + kk) * N + nb + np * 2);
            sBt[(np * 2) * 32 + kk]     = (u16)(uv & 0xffffu);
            sBt[(np * 2 + 1) * 32 + kk] = (u16)(uv >> 16);
        }
        if (kb + 32 < K) {                      // prefetch next K tiles into GL2
            __builtin_prefetch(A + (size_t)(mb + tid) * K + kb + 32, 0, 0);
            if (tid < 32) __builtin_prefetch(B + (size_t)(kb + 32 + tid) * N + nb, 0, 0);
        }
        __syncthreads();
        v16bf a0 = load_afrag16(sA, 32, w * 16 + (lane & 15), 0, lane);
        v16bf a1 = load_afrag16(sA, 32, 64 + w * 16 + (lane & 15), 0, lane);
#pragma unroll
        for (int t = 0; t < 4; ++t) {
            v16bf b = load_bfrag16(sBt, 32, t * 16 + (lane & 15), 0, lane);
            acc[0][t] = wmma_bf16(a0, b, acc[0][t]);
            acc[1][t] = wmma_bf16(a1, b, acc[1][t]);
        }
    }

#pragma unroll
    for (int m2 = 0; m2 < 2; ++m2) {
        int row0 = mb + m2 * 64 + w * 16 + ((lane & 16) ? 8 : 0);
#pragma unroll
        for (int t = 0; t < 4; ++t) {
            int col = nb + t * 16 + (lane & 15);
            float bv = bias[col];
#pragma unroll
            for (int r = 0; r < 8; ++r) {
                int row = row0 + r;
                float v = acc[m2][t][r] + bv;
                if constexpr (RESID) v += resid[(size_t)row * N + col];
                if constexpr (GELU)  v = 0.5f * v * (1.f + erff(v * 0.70710678118654752f));
                if constexpr (OUTF32) ((float*)Cout)[(size_t)row * N + col] = v;
                else                  ((u16*)Cout)[(size_t)row * N + col] = f2bf(v);
            }
        }
    }
}

// ---------- Fused sliding-window attention (K window staged by TDM) ----------
// grid: (S/32, NHEAD). block = 256 threads (8 waves).
// LDS: sQ 32x64 bf16 (4KB) | sK 768x64 bf16 -> reused as sVt 64x768 bf16 (96KB) | sS 32x768 f32 (96KB)
#define ATTN_LDS (4096 + 98304 + 98304)
__global__ __launch_bounds__(256) void attn_kernel(const u16* __restrict__ qb,
                                                   const u16* __restrict__ kb,
                                                   const u16* __restrict__ vb,
                                                   const int* __restrict__ amask,
                                                   u16* __restrict__ ab) {
    extern __shared__ char smem[];
    u16*   sQ  = (u16*)smem;                     // [32][64]
    u16*   sKV = (u16*)(smem + 4096);            // [768][64] K, then [64][768] V^T
    float* sS  = (float*)(smem + 4096 + 98304);  // [32][768]

    int tile = blockIdx.x, h = blockIdx.y;
    int tid = threadIdx.x, lane = tid & 31, w = tid >> 5;
    int qg0 = tile * 32;
    int c = tile >> 3;                // chunk of 256
    int kstart = (c - 1) * WWIN;      // first key position in window

    // ---- phase 1a: zero-fill leading OOB rows (first chunk only), load Q (scaled) ----
    if (kstart < 0) {
#pragma unroll
        for (int i = 0; i < 64; ++i) sKV[tid + i * 256] = 0;   // rows 0..255
    }
#pragma unroll
    for (int i = 0; i < 8; ++i) {
        int idx = tid + i * 256;
        int r = idx >> 6, d = idx & 63;
        float qv = bf2f(qb[(size_t)(qg0 + r) * HID + h * DHEAD + d]) * 0.125f;
        sQ[r * 64 + d] = f2bf(qv);
    }

    // ---- phase 1b: TDM 2-D tile load of the K window, global -> LDS ----
    // D# group0: count=1 | lds_addr | 57-bit global_addr | type=2 ("image")
    // D# group1: data_size=2B, tensor_dim0=64, tensor_dim1=S-row0 (trailing OOB rows read 0),
    //            tile_dim0=64, tile_dim1=768-skip, tensor_dim0_stride=HID
    if (w == 0) {
        int row0 = (kstart < 0) ? 0 : kstart;
        int skip = row0 - kstart;                                   // leading zero rows
        unsigned ldsaddr = (unsigned)(size_t)(void*)sKV + (unsigned)(skip * (DHEAD * 2));
        unsigned long long gaddr =
            (unsigned long long)(size_t)(kb + (size_t)row0 * HID + h * DHEAD);
        unsigned vrows = (unsigned)(S_LEN - row0);                  // tensor_dim1
        unsigned trows = (unsigned)(768 - skip);                    // tile_dim1
        su4 g0 = { rfl(1u),                                         // count=1, user mode
                   rfl(ldsaddr),
                   rfl((unsigned)gaddr),
                   rfl((unsigned)((gaddr >> 32) & 0x01FFFFFFu) | (2u << 30)) };
        su8 g1 = { rfl(0x00010000u),                                // data_size=2B
                   rfl((unsigned)DHEAD << 16),                      // tensor_dim0=64 (lo16)
                   rfl((vrows & 0xFFFFu) << 16),                    // dim0 hi | tensor_dim1 lo
                   rfl(((vrows >> 16) & 0xFFFFu) | ((unsigned)DHEAD << 16)),  // dim1 hi | tile_dim0
                   rfl(trows & 0xFFFFu),                            // tile_dim1 | tile_dim2=0
                   rfl((unsigned)HID),                              // tensor_dim0_stride lo32
                   rfl(0u),                                         // stride hi | dim1_stride lo
                   rfl(0u) };
        asm volatile("tensor_load_to_lds %0, %1" :: "s"(g0), "s"(g1) : "memory");
        __builtin_amdgcn_s_wait_tensorcnt(0);
    }
    __syncthreads();

    // ---- phase 2: scores = Q @ K^T (16x16 tiles, K-dim = 64 -> 2 WMMAs) ----
    {
        int qt = w & 1;           // 2 query tiles of 16
        int ksec = w >> 1;        // 4 key sections of 192
        for (int t = 0; t < 12; ++t) {
            int kb16 = ksec * 192 + t * 16;
            v8f sc = {};
#pragma unroll
            for (int db = 0; db < 64; db += 32) {
                v16bf a = load_afrag16(sQ, 64, qt * 16 + (lane & 15), db, lane);
                v16bf b = load_bfrag16(sKV, 64, kb16 + (lane & 15), db, lane);
                sc = wmma_bf16(a, b, sc);
            }
            int row0 = qt * 16 + ((lane & 16) ? 8 : 0);
            int col = kb16 + (lane & 15);
#pragma unroll
            for (int r = 0; r < 8; ++r) sS[(row0 + r) * 768 + col] = sc[r];
        }
    }
    __syncthreads();

    // ---- phase 3a: masked softmax over 768 keys (8 lanes per row) ----
    {
        int row = tid >> 3;              // 0..31
        int sub = tid & 7;
        int qg = qg0 + row;
        float mx = NEGV;
        for (int i = 0; i < 96; ++i) {
            int j = sub + i * 8;
            int kpos = kstart + j;
            bool ok = (kpos >= 0) && (kpos < S_LEN) &&
                      (abs(kpos - qg) <= WWIN) && (amask[kpos] != 0);
            float v = ok ? sS[row * 768 + j] : NEGV;
            mx = fmaxf(mx, v);
        }
        for (int m = 1; m < 8; m <<= 1) mx = fmaxf(mx, __shfl_xor(mx, m));
        float sum = 0.f;
        for (int i = 0; i < 96; ++i) {
            int j = sub + i * 8;
            int kpos = kstart + j;
            bool ok = (kpos >= 0) && (kpos < S_LEN) &&
                      (abs(kpos - qg) <= WWIN) && (amask[kpos] != 0);
            float v = ok ? sS[row * 768 + j] : NEGV;
            sum += __expf(v - mx);
        }
        for (int m = 1; m < 8; m <<= 1) sum += __shfl_xor(sum, m);
        float inv = 1.f / sum;
        for (int i = 0; i < 96; ++i) {
            int j = sub + i * 8;
            int kpos = kstart + j;
            bool ok = (kpos >= 0) && (kpos < S_LEN) &&
                      (abs(kpos - qg) <= WWIN) && (amask[kpos] != 0);
            float v = ok ? sS[row * 768 + j] : NEGV;
            sS[row * 768 + j] = __expf(v - mx) * inv;
        }
    }
    // ---- phase 3b: load V transposed into the K region ----
    for (int i = 0; i < 192; ++i) {
        int idx = tid + i * 256;
        int j = idx >> 6, d = idx & 63;
        int kpos = kstart + j;
        u16 vv = 0;
        if (kpos >= 0 && kpos < S_LEN) vv = vb[(size_t)kpos * HID + h * DHEAD + d];
        sKV[d * 768 + j] = vv;     // V^T : [d][key]
    }
    __syncthreads();

    // ---- phase 4: out = P @ V (each wave: one 16x16 output tile) ----
    {
        int qt = w & 1;
        int dt = w >> 1;           // 4 d-tiles of 16
        v8f oc = {};
        for (int ks = 0; ks < 768; ks += 32) {
            v16bf a = load_afrag_f32(&sS[(qt * 16 + (lane & 15)) * 768], ks, lane);
            v16bf b = load_bfrag16(sKV, 768, dt * 16 + (lane & 15), ks, lane);
            oc = wmma_bf16(a, b, oc);
        }
        int row0 = qt * 16 + ((lane & 16) ? 8 : 0);
        int d = dt * 16 + (lane & 15);
#pragma unroll
        for (int r = 0; r < 8; ++r) {
            int s = qg0 + row0 + r;
            ab[(size_t)s * HID + h * DHEAD + d] = f2bf(oc[r]);
        }
    }
}

// ---------- classification head: out[5] = x[0,:] @ head_w + head_b ----------
__global__ __launch_bounds__(256) void head_kernel(const float* __restrict__ x,
                                                   const float* __restrict__ hw,
                                                   const float* __restrict__ hb,
                                                   float* __restrict__ out) {
    __shared__ float red[256];
    int tid = threadIdx.x;
    for (int l = 0; l < 5; ++l) {
        float p = 0.f;
        for (int d = tid; d < HID; d += 256) p += x[d] * hw[(size_t)d * 5 + l];
        red[tid] = p;
        __syncthreads();
        for (int off = 128; off > 0; off >>= 1) {
            if (tid < off) red[tid] += red[tid + off];
            __syncthreads();
        }
        if (tid == 0) out[l] = red[0] + hb[l];
        __syncthreads();
    }
}

extern "C" void kernel_launch(void* const* d_in, const int* in_sizes, int n_in,
                              void* d_out, int out_size, void* d_ws, size_t ws_size,
                              hipStream_t stream) {
    (void)in_sizes; (void)n_in; (void)out_size; (void)ws_size;
    auto F = [&](int i) { return (const float*)d_in[i]; };
    const float* word_emb = F(0);
    const float* pos_emb  = F(1);
    const float* emb_g    = F(2);
    const float* emb_b    = F(3);
    const float* head_w   = F(4);
    const float* head_b   = F(5);
    const int* ids   = (const int*)d_in[38];
    const int* amask = (const int*)d_in[39];

    char* ws = (char*)d_ws;
    size_t off = 0;
    auto carve = [&](size_t bytes) { void* p = ws + off; off += (bytes + 255) & ~(size_t)255; return p; };
    const size_t SH = (size_t)S_LEN * HID;
    u16*   XB = (u16*)  carve(SH * 2);                  // x bf16
    float* XF = (float*)carve(SH * 4);                  // x f32
    float* TF = (float*)carve(SH * 4);                  // pre-LN temp
    u16*   QB = (u16*)  carve(SH * 2);
    u16*   KB = (u16*)  carve(SH * 2);
    u16*   VB = (u16*)  carve(SH * 2);
    u16*   AB = (u16*)  carve(SH * 2);                  // attention out bf16
    u16*   HB = (u16*)  carve((size_t)S_LEN * FFDIM * 2);
    u16*   WB = (u16*)  carve((size_t)FFDIM * HID * 2); // weight staging (max)

    auto cvt = [&](const float* src, u16* dst, int n) {
        f2bf_kernel<<<(n + 255) / 256, 256, 0, stream>>>(src, dst, n);
    };
    auto gemm_plain = [&](const u16* A, const u16* Bw, const float* bias, u16* Cb,
                          int M, int N, int K) {
        gemm_wmma<false, false, false><<<dim3(M / 128, N / 64), 128, 0, stream>>>(
            A, Bw, bias, nullptr, (void*)Cb, M, N, K);
    };
    auto gemm_resid = [&](const u16* A, const u16* Bw, const float* bias, const float* resid,
                          float* Cf, int M, int N, int K) {
        gemm_wmma<true, false, true><<<dim3(M / 128, N / 64), 128, 0, stream>>>(
            A, Bw, bias, resid, (void*)Cf, M, N, K);
    };
    auto gemm_gelu = [&](const u16* A, const u16* Bw, const float* bias, u16* Cb,
                         int M, int N, int K) {
        gemm_wmma<false, true, false><<<dim3(M / 128, N / 64), 128, 0, stream>>>(
            A, Bw, bias, nullptr, (void*)Cb, M, N, K);
    };

    // Embedding + LN
    embed_kernel<<<S_LEN, 256, 0, stream>>>(ids, word_emb, pos_emb, TF);
    ln_kernel<<<S_LEN, 256, 0, stream>>>(TF, emb_g, emb_b, XF, XB);

    for (int l = 0; l < 2; ++l) {
        int b = 6 + l * 16;
        const float *wq = F(b + 0),  *bq = F(b + 1);
        const float *wk = F(b + 2),  *bk = F(b + 3);
        const float *wv = F(b + 4),  *bv = F(b + 5);
        const float *wo = F(b + 6),  *bo = F(b + 7);
        const float *g1 = F(b + 8),  *be1 = F(b + 9);
        const float *w1 = F(b + 10), *b1 = F(b + 11);
        const float *w2 = F(b + 12), *b2 = F(b + 13);
        const float *g2 = F(b + 14), *be2 = F(b + 15);

        cvt(wq, WB, HID * HID); gemm_plain(XB, WB, bq, QB, S_LEN, HID, HID);
        cvt(wk, WB, HID * HID); gemm_plain(XB, WB, bk, KB, S_LEN, HID, HID);
        cvt(wv, WB, HID * HID); gemm_plain(XB, WB, bv, VB, S_LEN, HID, HID);

        attn_kernel<<<dim3(S_LEN / 32, NHEAD), 256, ATTN_LDS, stream>>>(QB, KB, VB, amask, AB);

        cvt(wo, WB, HID * HID); gemm_resid(AB, WB, bo, XF, TF, S_LEN, HID, HID);
        ln_kernel<<<S_LEN, 256, 0, stream>>>(TF, g1, be1, XF, XB);

        cvt(w1, WB, HID * FFDIM); gemm_gelu(XB, WB, b1, HB, S_LEN, FFDIM, HID);
        cvt(w2, WB, FFDIM * HID); gemm_resid(HB, WB, b2, XF, TF, S_LEN, HID, FFDIM);
        ln_kernel<<<S_LEN, 256, 0, stream>>>(TF, g2, be2, XF, XB);
    }

    head_kernel<<<1, 256, 0, stream>>>(XF, head_w, head_b, (float*)d_out);
}
